// Trnsform_target_65996467470920
// MI455X (gfx1250) — compile-verified, matrix operand order checked by reading, not run on hardware
//
#include <hip/hip_runtime.h>
#include <stdint.h>

typedef __attribute__((ext_vector_type(16))) _Float16 v16h;
typedef __attribute__((ext_vector_type(8)))  float    v8f;

#define B_    16
#define P_    8732
#define C_    81
#define K_    200
#define NOBJ_ 16
#define NCLS_ 80          // classes 1..80 (class 0 skipped)
#define M_    (NCLS_*K_)  // 16000 candidates per image
#define NEGV  (-1e30f)
#define CONF_TH 0.01f
#define NMS_TH  0.45f
#define OV_TH   0.5f
#define V0_  0.1f
#define V1_  0.2f

#define NTILE_ ((K_ + 15) / 16)          // 13 tiles of 16 over the 200 boxes

__device__ __forceinline__ void wait_asynccnt0() {
#if __has_builtin(__builtin_amdgcn_s_wait_asynccnt)
    __builtin_amdgcn_s_wait_asynccnt(0);
#else
    asm volatile("s_wait_asynccnt 0" ::: "memory");
#endif
}

// ---------------------------------------------------------------------------
// Kernel 1: match.  One block per image.  16x8732 Jaccard with the union
// outer-sum (areaT_i + areaP_j) computed by v_wmma_f32_16x16x32_f16 as a
// rank-2 matmul; argmax over priors (per truth, LDS u64 atomicMax) and over
// truths (per prior); then best-prior override + encode -> loc_t / conf_t.
// ---------------------------------------------------------------------------
__global__ __launch_bounds__(256) void match_kernel(
    const float* __restrict__ prior, const float* __restrict__ targets,
    float* __restrict__ loc_t, float* __restrict__ conf_t,
    float* __restrict__ bt_ov, int* __restrict__ bt_idx)
{
    const int b    = blockIdx.x;
    const int tid  = threadIdx.x;
    const int wave = tid >> 5;
    const int lane = tid & 31;

    __shared__ float tb[NOBJ_][4];
    __shared__ float tarea[NOBJ_];
    __shared__ float tlab[NOBJ_];
    __shared__ float pb[8][16][4];
    __shared__ float pa[8][16];
    __shared__ float wt[8][16][16];
    __shared__ unsigned long long truthBest[NOBJ_];

    if (tid < NOBJ_) {
        const float* t = targets + ((size_t)b * NOBJ_ + tid) * 5;
        tb[tid][0] = t[0]; tb[tid][1] = t[1]; tb[tid][2] = t[2]; tb[tid][3] = t[3];
        tarea[tid] = (t[2] - t[0]) * (t[3] - t[1]);
        tlab[tid]  = t[4];
        truthBest[tid] = 0ull;
    }
    __syncthreads();

    const int NT = (P_ + 15) / 16;          // 546 prior tiles
    for (int base = 0; base < NT; base += 8) {
        const int  tile = base + wave;
        const bool act  = (tile < NT);

        // stage 16 priors (point form + area) per active wave
        if (act && lane < 16) {
            int   p = tile * 16 + lane;
            float bx0 = 0.f, by0 = 0.f, bx1 = 0.f, by1 = 0.f, area = 0.f;
            if (p < P_) {
                float cx = prior[p * 4 + 0], cy = prior[p * 4 + 1];
                float w_ = prior[p * 4 + 2], h_ = prior[p * 4 + 3];
                bx0 = cx - w_ * 0.5f; by0 = cy - h_ * 0.5f;
                bx1 = cx + w_ * 0.5f; by1 = cy + h_ * 0.5f;
                area = w_ * h_;
            }
            pb[wave][lane][0] = bx0; pb[wave][lane][1] = by0;
            pb[wave][lane][2] = bx1; pb[wave][lane][3] = by1;
            pa[wave][lane]    = area;
        }
        __syncthreads();

        if (act) {
            // A = [areaT | 1 | 0...], B = [1 ; areaP ; 0...] -> D = aT_i + aP_j
            v16h av = {}; v16h bv = {};
            if (lane < 16) {
                av[0] = (_Float16)tarea[lane];
                av[1] = (_Float16)1.0f;
                bv[0] = (_Float16)1.0f;
                bv[1] = (_Float16)pa[wave][lane];
            }
            v8f cv = {};
            v8f dv = __builtin_amdgcn_wmma_f32_16x16x32_f16(
                false, av, false, bv, (short)0, cv, false, false);

            const int Ni = lane & 15;
            const int pp = tile * 16 + Ni;
#pragma unroll
            for (int r = 0; r < 8; ++r) {
                const int Mi = r + ((lane >= 16) ? 8 : 0);
                float iou = -1.0f;
                if (pp < P_) {
                    float lx = fmaxf(tb[Mi][0], pb[wave][Ni][0]);
                    float ly = fmaxf(tb[Mi][1], pb[wave][Ni][1]);
                    float rx = fminf(tb[Mi][2], pb[wave][Ni][2]);
                    float ry = fminf(tb[Mi][3], pb[wave][Ni][3]);
                    float iw = fmaxf(rx - lx, 0.f), ih = fmaxf(ry - ly, 0.f);
                    float inter = iw * ih;
                    iou = inter * __builtin_amdgcn_rcpf(dv[r] - inter);
                }
                wt[wave][Mi][Ni] = iou;
            }
        }
        __syncthreads();

        if (tid < 128) {
            // per-truth running argmax over priors (tie -> lowest prior idx)
            const int w2 = tid >> 4, m2 = tid & 15;
            const int tile2 = base + w2;
            if (tile2 < NT) {
                unsigned long long lb = 0ull;
                for (int N = 0; N < 16; ++N) {
                    int p2 = tile2 * 16 + N;
                    if (p2 >= P_) break;
                    float v = wt[w2][m2][N];
                    if (v < 0.f) continue;
                    unsigned long long key =
                        ((unsigned long long)__float_as_uint(v) << 32) |
                        (unsigned long long)(0xFFFFFFFFu - (unsigned)p2);
                    if (key > lb) lb = key;
                }
                if (lb) atomicMax(&truthBest[m2], lb);
            }
        } else {
            // per-prior argmax over truths (tie -> lowest truth idx)
            const int idx = tid - 128;
            const int w2 = idx >> 4, n2 = idx & 15;
            const int tile2 = base + w2;
            const int p2 = tile2 * 16 + n2;
            if (tile2 < NT && p2 < P_) {
                float bv2 = wt[w2][0][n2]; int ba2 = 0;
                for (int Mi = 1; Mi < 16; ++Mi) {
                    float v = wt[w2][Mi][n2];
                    if (v > bv2) { bv2 = v; ba2 = Mi; }
                }
                bt_ov [(size_t)b * P_ + p2] = bv2;
                bt_idx[(size_t)b * P_ + p2] = ba2;
            }
        }
        __syncthreads();
    }

    // best-prior override, sequential so duplicate priors resolve last-wins
    if (tid == 0) {
        for (int t = 0; t < NOBJ_; ++t) {
            unsigned int p = 0xFFFFFFFFu - (unsigned int)(truthBest[t] & 0xFFFFFFFFull);
            bt_ov [(size_t)b * P_ + p] = 2.0f;
            bt_idx[(size_t)b * P_ + p] = t;
        }
    }
    __syncthreads();

    // encode matched boxes
    for (int p = tid; p < P_; p += 256) {
        int   ti = bt_idx[(size_t)b * P_ + p];
        float ov = bt_ov [(size_t)b * P_ + p];
        float mn0 = tb[ti][0], mn1 = tb[ti][1], mx0 = tb[ti][2], mx1 = tb[ti][3];
        float cx = prior[p * 4 + 0], cy = prior[p * 4 + 1];
        float pw = prior[p * 4 + 2], ph = prior[p * 4 + 3];
        size_t o = ((size_t)b * P_ + p) * 4;
        loc_t[o + 0] = ((mn0 + mx0) * 0.5f - cx) / (V0_ * pw);
        loc_t[o + 1] = ((mn1 + mx1) * 0.5f - cy) / (V0_ * ph);
        loc_t[o + 2] = logf((mx0 - mn0) / pw) / V1_;
        loc_t[o + 3] = logf((mx1 - mn1) / ph) / V1_;
        conf_t[(size_t)b * P_ + p] = (ov < OV_TH) ? 0.f : (tlab[ti] + 1.f);
    }
}

// ---------------------------------------------------------------------------
// Kernel 2: decode all B*P boxes.
// ---------------------------------------------------------------------------
__global__ __launch_bounds__(256) void decode_kernel(
    const float* __restrict__ loc, const float* __restrict__ prior,
    float* __restrict__ dec)
{
    int i = blockIdx.x * 256 + threadIdx.x;
    if (i >= B_ * P_) return;
    int p = i % P_;
    const float* l = loc + (size_t)i * 4;
    float cx = prior[p * 4 + 0] + l[0] * V0_ * prior[p * 4 + 2];
    float cy = prior[p * 4 + 1] + l[1] * V0_ * prior[p * 4 + 3];
    float w_ = prior[p * 4 + 2] * expf(l[2] * V1_);
    float h_ = prior[p * 4 + 3] * expf(l[3] * V1_);
    float* d = dec + (size_t)i * 4;
    d[0] = cx - w_ * 0.5f; d[1] = cy - h_ * 0.5f;
    d[2] = cx + w_ * 0.5f; d[3] = cy + h_ * 0.5f;
}

// ---------------------------------------------------------------------------
// Kernel 3: per (image,class) top-200 + NMS.  1280 blocks.
//  - 200 masked-argmax sweeps (conf is L2-resident, 45MB << 192MB L2)
//  - chosen boxes gathered into LDS with global_load_async_to_lds_b128
//  - 200x200 IoU adjacency: 13x13 WMMA tiles (union via rank-2 outer-sum),
//    threshold test inter > T*union (division-free), bits via wave32 ballot
//  - greedy lax.scan collapses to a serial 200x7-word bit-OR
// ---------------------------------------------------------------------------
__global__ __launch_bounds__(256) void nms_kernel(
    const float* __restrict__ conf, const float* __restrict__ dec,
    int* __restrict__ topi, float* __restrict__ tops, int* __restrict__ keepw)
{
    const int blk  = blockIdx.x;
    const int b    = blk / NCLS_;
    const int c    = 1 + (blk % NCLS_);
    const int tid  = threadIdx.x;
    const int wave = tid >> 5;
    const int lane = tid & 31;

    __shared__ unsigned int        bitmap[(P_ + 31) / 32];
    __shared__ unsigned long long  red[256];
    __shared__ int    chIdx[K_];
    __shared__ float  chSc[K_];
    __shared__ __align__(16) float bx[K_][4];
    __shared__ float  ba[K_];
    __shared__ int    keepA[K_];
    __shared__ unsigned short adj[NTILE_ * 16][NTILE_ + 1];   // 208 x 14 bitmap rows

    for (int i = tid; i < (P_ + 31) / 32; i += 256) bitmap[i] = 0u;
    __syncthreads();

    const float* scp = conf + c;   // stride C_ over rows (b*P + p)
    __builtin_prefetch(&conf[((size_t)b * P_) * C_ + c], 0, 1);

    // 200 masked argmax passes (key packs ordered-float score + ~p tiebreak)
    for (int it = 0; it < K_; ++it) {
        unsigned long long best = 0ull;
        for (int p = tid; p < P_; p += 256) {
            if (bitmap[p >> 5] & (1u << (p & 31))) continue;
            float s = scp[(size_t)(b * P_ + p) * C_];
            float m = (s > CONF_TH) ? s : NEGV;
            unsigned int ub = __float_as_uint(m);
            ub ^= (ub & 0x80000000u) ? 0xFFFFFFFFu : 0x80000000u;
            unsigned long long key = ((unsigned long long)ub << 32) |
                                     (unsigned long long)(0xFFFFFFFFu - (unsigned)p);
            if (key > best) best = key;
        }
        red[tid] = best;
        __syncthreads();
        for (int s2 = 128; s2 > 0; s2 >>= 1) {
            if (tid < s2 && red[tid + s2] > red[tid]) red[tid] = red[tid + s2];
            __syncthreads();
        }
        if (tid == 0) {
            unsigned long long k = red[0];
            unsigned int p  = 0xFFFFFFFFu - (unsigned int)(k & 0xFFFFFFFFull);
            unsigned int ub = (unsigned int)(k >> 32);
            ub ^= (ub & 0x80000000u) ? 0x80000000u : 0xFFFFFFFFu;
            chIdx[it] = (int)p;
            chSc[it]  = __uint_as_float(ub);
            bitmap[p >> 5] |= (1u << (p & 31));
        }
        __syncthreads();
    }

    // async gather of the 200 chosen decoded boxes directly into LDS
    if (tid < K_) {
        unsigned ldsa = (unsigned)(uintptr_t)(&bx[tid][0]);
        unsigned long long ga =
            (unsigned long long)(uintptr_t)(dec + ((size_t)b * P_ + chIdx[tid]) * 4);
        asm volatile("global_load_async_to_lds_b128 %0, %1, off"
                     :: "v"(ldsa), "v"(ga) : "memory");
    }
    wait_asynccnt0();
    __syncthreads();
    if (tid < K_) {
        ba[tid]    = (bx[tid][2] - bx[tid][0]) * (bx[tid][3] - bx[tid][1]);
        keepA[tid] = 0;
    }
    __syncthreads();

    // adjacency bitmap: 13x13 WMMA tiles of the 200x200 IoU matrix
    for (int t = wave; t < NTILE_ * NTILE_; t += 8) {
        const int ti = t / NTILE_, tj = t % NTILE_;
        const int rowb = ti * 16, colb = tj * 16;
        v16h av = {}; v16h bv = {};
        if (lane < 16) {
            int rr = rowb + lane, cc = colb + lane;
            av[0] = (_Float16)((rr < K_) ? ba[rr] : 0.f);
            av[1] = (_Float16)1.0f;
            bv[0] = (_Float16)1.0f;
            bv[1] = (_Float16)((cc < K_) ? ba[cc] : 0.f);
        }
        v8f cv = {};
        v8f dv = __builtin_amdgcn_wmma_f32_16x16x32_f16(
            false, av, false, bv, (short)0, cv, false, false);

        const int col = colb + (lane & 15);
#pragma unroll
        for (int r = 0; r < 8; ++r) {
            const int row = rowb + r + ((lane >= 16) ? 8 : 0);
            bool bit = false;
            if (row < K_ && col < K_) {
                float lx = fmaxf(bx[row][0], bx[col][0]);
                float ly = fmaxf(bx[row][1], bx[col][1]);
                float rx = fminf(bx[row][2], bx[col][2]);
                float ry = fminf(bx[row][3], bx[col][3]);
                float iw = fmaxf(rx - lx, 0.f), ih = fmaxf(ry - ly, 0.f);
                float inter = iw * ih;
                bit = inter > NMS_TH * (dv[r] - inter);   // iou > T
            }
            unsigned long long m = __ballot(bit);
            if (lane == 0) {
                int r0 = rowb + r;
                if (r0 < K_)     adj[r0][tj]     = (unsigned short)(m & 0xFFFFull);
                int r1 = rowb + r + 8;
                if (r1 < K_)     adj[r1][tj]     = (unsigned short)((m >> 16) & 0xFFFFull);
            }
        }
    }
    __syncthreads();

    // greedy suppression scan (exact lax.scan semantics) as serial bit ops
    if (tid == 0) {
        unsigned int suppw[7] = {0u, 0u, 0u, 0u, 0u, 0u, 0u};
        for (int k = 0; k < K_; ++k) {
            int kf = (chSc[k] > CONF_TH) && !((suppw[k >> 5] >> (k & 31)) & 1u);
            keepA[k] = kf;
            if (kf) {
                for (int w2 = 0; w2 < 6; ++w2)
                    suppw[w2] |= (unsigned)adj[k][2 * w2] |
                                 ((unsigned)adj[k][2 * w2 + 1] << 16);
                suppw[6] |= (unsigned)adj[k][12];
            }
        }
    }
    __syncthreads();

    size_t ob = (size_t)blk * K_;
    if (tid < K_) {
        topi [ob + tid] = chIdx[tid];
        tops [ob + tid] = chSc[tid];
        keepw[ob + tid] = keepA[tid];
    }
}

// ---------------------------------------------------------------------------
// Kernel 4: per image — first-occurrence dedup by prior, final top-200, gather.
// ---------------------------------------------------------------------------
__global__ __launch_bounds__(256) void select_kernel(
    const float* __restrict__ loc_data, const float* __restrict__ conf,
    const float* __restrict__ dec, const float* __restrict__ loc_t,
    const float* __restrict__ conf_t, const int* __restrict__ topi,
    const float* __restrict__ tops, const int* __restrict__ keepw,
    int* __restrict__ first, float* __restrict__ out)
{
    const int b   = blockIdx.x;
    const int tid = threadIdx.x;
    const int BIG = M_ + 1;

    __shared__ unsigned int       bitmap[(M_ + 31) / 32];
    __shared__ unsigned long long red[256];
    __shared__ int   selM[K_];
    __shared__ float selV[K_];

    for (int p = tid; p < P_; p += 256) first[(size_t)b * P_ + p] = BIG;
    for (int i = tid; i < (M_ + 31) / 32; i += 256) bitmap[i] = 0u;
    __syncthreads();

    const size_t mb = (size_t)b * M_;
    for (int m = tid; m < M_; m += 256)
        if (keepw[mb + m]) atomicMin(&first[(size_t)b * P_ + topi[mb + m]], m);
    __syncthreads();

    for (int it = 0; it < K_; ++it) {
        unsigned long long best = 0ull;
        for (int m = tid; m < M_; m += 256) {
            if (bitmap[m >> 5] & (1u << (m & 31))) continue;
            int  pid = topi[mb + m];
            bool sel = keepw[mb + m] && (m == first[(size_t)b * P_ + pid]);
            float v  = sel ? tops[mb + m] : NEGV;
            unsigned int ub = __float_as_uint(v);
            ub ^= (ub & 0x80000000u) ? 0xFFFFFFFFu : 0x80000000u;
            unsigned long long key = ((unsigned long long)ub << 32) |
                                     (unsigned long long)(0xFFFFFFFFu - (unsigned)m);
            if (key > best) best = key;
        }
        red[tid] = best;
        __syncthreads();
        for (int s2 = 128; s2 > 0; s2 >>= 1) {
            if (tid < s2 && red[tid + s2] > red[tid]) red[tid] = red[tid + s2];
            __syncthreads();
        }
        if (tid == 0) {
            unsigned long long k = red[0];
            unsigned int m  = 0xFFFFFFFFu - (unsigned int)(k & 0xFFFFFFFFull);
            unsigned int ub = (unsigned int)(k >> 32);
            ub ^= (ub & 0x80000000u) ? 0x80000000u : 0xFFFFFFFFu;
            selM[it] = (int)m;
            selV[it] = __uint_as_float(ub);
            bitmap[m >> 5] |= (1u << (m & 31));
        }
        __syncthreads();
    }

    // gather 95-column result split into the 5 output tensors
    const int COLS = 5 + 4 + C_ + 4 + 1;            // 95
    float* rois   = out;
    float* oloc   = out   + (size_t)B_ * K_ * 5;
    float* ocls   = oloc  + (size_t)B_ * K_ * 4;
    float* oloct  = ocls  + (size_t)B_ * K_ * C_;
    float* oconft = oloct + (size_t)B_ * K_ * 4;

    for (int e = tid; e < K_ * COLS; e += 256) {
        int k = e / COLS, col = e % COLS;
        float v  = (selV[k] > NEGV * 0.5f) ? 1.f : 0.f;
        int   sp = topi[mb + selM[k]];
        size_t row = (size_t)b * K_ + k;
        if (col == 0)            rois  [row * 5]                = v * (float)b;
        else if (col < 5)        rois  [row * 5  + col]         = v * dec     [((size_t)b * P_ + sp) * 4 + (col - 1)];
        else if (col < 9)        oloc  [row * 4  + (col - 5)]   = v * loc_data[((size_t)b * P_ + sp) * 4 + (col - 5)];
        else if (col < 9 + C_)   ocls  [row * C_ + (col - 9)]   = v * conf    [((size_t)b * P_ + sp) * C_ + (col - 9)];
        else if (col < 13 + C_)  oloct [row * 4  + (col - 9 - C_)] = v * loc_t[((size_t)b * P_ + sp) * 4 + (col - 9 - C_)];
        else                     oconft[row]                    = v * conf_t  [(size_t)b * P_ + sp];
    }
}

// ---------------------------------------------------------------------------
extern "C" void kernel_launch(void* const* d_in, const int* in_sizes, int n_in,
                              void* d_out, int out_size, void* d_ws, size_t ws_size,
                              hipStream_t stream)
{
    (void)in_sizes; (void)n_in; (void)out_size; (void)ws_size;
    const float* loc_data  = (const float*)d_in[0];
    const float* conf_data = (const float*)d_in[1];
    const float* prior     = (const float*)d_in[2];
    const float* targets   = (const float*)d_in[3];

    char* w = (char*)d_ws;
    float* dec    = (float*)w; w += (size_t)B_ * P_ * 4 * sizeof(float);
    float* loc_t  = (float*)w; w += (size_t)B_ * P_ * 4 * sizeof(float);
    float* conf_t = (float*)w; w += (size_t)B_ * P_ * sizeof(float);
    float* bt_ov  = (float*)w; w += (size_t)B_ * P_ * sizeof(float);
    int*   bt_idx = (int*)  w; w += (size_t)B_ * P_ * sizeof(int);
    int*   topi   = (int*)  w; w += (size_t)B_ * M_ * sizeof(int);
    float* tops   = (float*)w; w += (size_t)B_ * M_ * sizeof(float);
    int*   keepw  = (int*)  w; w += (size_t)B_ * M_ * sizeof(int);
    int*   first  = (int*)  w; w += (size_t)B_ * P_ * sizeof(int);

    match_kernel <<<B_, 256, 0, stream>>>(prior, targets, loc_t, conf_t, bt_ov, bt_idx);
    decode_kernel<<<(B_ * P_ + 255) / 256, 256, 0, stream>>>(loc_data, prior, dec);
    nms_kernel   <<<B_ * NCLS_, 256, 0, stream>>>(conf_data, dec, topi, tops, keepw);
    select_kernel<<<B_, 256, 0, stream>>>(loc_data, conf_data, dec, loc_t, conf_t,
                                          topi, tops, keepw, first, (float*)d_out);
}
